// QSFormer_52226802319471
// MI455X (gfx1250) — matrix operand order repaired and checked by reading, not compiled, not run on hardware
//
#include <hip/hip_runtime.h>

// ---------------- problem constants ----------------
#define WAYN   16
#define NQN    240
#define CN     640
#define HWN    49
#define NHN    8
#define HDN    80
#define FFNN   2048
#define QCH    48          // query chunk (49*48 = 2352 rows, multiple of 16)
#define QROWS  (HWN*QCH)   // 2352

typedef __attribute__((ext_vector_type(16))) _Float16 v16h;
typedef __attribute__((ext_vector_type(8)))  float    v8f;

static inline int cdiv(int a, int b) { return (a + b - 1) / b; }

// =====================================================================
// fp32 -> (hi, lo) f16 planes.  a = hi + lo to ~22 mantissa bits; the
// compensated WMMA product a*b ~ al*bh + ah*bl + ah*bh recovers near-fp32
// GEMM accuracy while running on the f16 matrix pipe.
// =====================================================================
__global__ void k_cvt_hilo(const float* __restrict__ X,
                           _Float16* __restrict__ hi, _Float16* __restrict__ lo, int n) {
  int i = blockIdx.x * blockDim.x + threadIdx.x;
  if (i >= n) return;
  float a = X[i];
  _Float16 h = (_Float16)a;
  hi[i] = h;
  lo[i] = (_Float16)(a - (float)h);
}

// =====================================================================
// GEMM on pre-split f16 planes: C[M,N] = act( A[M,K] * W[N,K]^T + bias[N] )
// One wave per 16x16 tile. All operand fragments are contiguous 16B runs
// in the ISA layout, so the K-loop is 8x global_load_b128 + 3x WMMA.
// Requires M%16==0, N%16==0, K%32==0 (true for every GEMM in this model).
// =====================================================================
__global__ void gemm_wmma_pre(const _Float16* __restrict__ Ahi, const _Float16* __restrict__ Alo,
                              const _Float16* __restrict__ Whi, const _Float16* __restrict__ Wlo,
                              const float* __restrict__ bias,
                              float* __restrict__ Cm, int ldc,
                              int M, int N, int K, int act) {
  const int lane   = threadIdx.x & 31;
  const int wave   = threadIdx.x >> 5;
  const int nTiles = N >> 4;
  const int tile   = blockIdx.x * (blockDim.x >> 5) + wave;
  const int mT     = tile / nTiles;
  const int nT     = tile - mT * nTiles;
  if (mT * 16 >= M) return;   // wave-uniform: EXEC stays all-ones for WMMA

  const int half = lane >> 4;
  const int l15  = lane & 15;
  // A layout (ISA 7.12.2, 16-bit A 16x32): per lane-half, K in two
  // contiguous 8-runs {8h..8h+7} and {16+8h..16+8h+7}.
  const _Float16* aHi = Ahi + (size_t)(mT * 16 + l15) * K + 8 * half;
  const _Float16* aLo = Alo + (size_t)(mT * 16 + l15) * K + 8 * half;
  // B layout (same as A): per lane-half, K in two contiguous 8-runs
  // {8h..8h+7} and {16+8h..16+8h+7}.  W row n == B column n.
  const _Float16* wHi = Whi + (size_t)(nT * 16 + l15) * K + 8 * half;
  const _Float16* wLo = Wlo + (size_t)(nT * 16 + l15) * K + 8 * half;

  v8f acc = {};
  for (int k = 0; k < K; k += 32) {
    union { v16h h; uint4 q[2]; } ah, al, bh, bl;
    ah.q[0] = *(const uint4*)(aHi + k);
    ah.q[1] = *(const uint4*)(aHi + k + 16);
    al.q[0] = *(const uint4*)(aLo + k);
    al.q[1] = *(const uint4*)(aLo + k + 16);
    bh.q[0] = *(const uint4*)(wHi + k);
    bh.q[1] = *(const uint4*)(wHi + k + 16);
    bl.q[0] = *(const uint4*)(wLo + k);
    bl.q[1] = *(const uint4*)(wLo + k + 16);
    acc = __builtin_amdgcn_wmma_f32_16x16x32_f16(false, al.h, false, bh.h, (short)0, acc, false, false);
    acc = __builtin_amdgcn_wmma_f32_16x16x32_f16(false, ah.h, false, bl.h, (short)0, acc, false, false);
    acc = __builtin_amdgcn_wmma_f32_16x16x32_f16(false, ah.h, false, bh.h, (short)0, acc, false, false);
  }
  // C layout: VGPR i -> row (i + 8*half), lane%16 -> col
  const int col  = nT * 16 + l15;
  const int row0 = mT * 16 + 8 * half;
  const float bz = bias ? bias[col] : 0.0f;
  #pragma unroll
  for (int i = 0; i < 8; ++i) {
    float v = acc[i] + bz;
    if (act) v = fmaxf(v, 0.0f);
    Cm[(size_t)(row0 + i) * ldc + col] = v;
  }
}

struct GemmScratch { _Float16 *ahi, *alo, *whi, *wlo; };

static void launch_gemm_pre(const _Float16* ahi, const _Float16* alo,
                            const _Float16* whi, const _Float16* wlo,
                            const float* bias, float* C, int ldc,
                            int M, int N, int K, int act, hipStream_t s) {
  int tiles = (M / 16) * (N / 16);
  gemm_wmma_pre<<<cdiv(tiles, 4), 128, 0, s>>>(ahi, alo, whi, wlo, bias, C, ldc, M, N, K, act);
}

// convert A and W (both row-length == K), then GEMM
static void launch_gemm(const float* A, const float* W, const float* bias,
                        float* C, int ldc, int M, int N, int K, int act,
                        const GemmScratch& sc, hipStream_t s) {
  int na = M * K, nw = N * K;
  k_cvt_hilo<<<cdiv(na, 256), 256, 0, s>>>(A, sc.ahi, sc.alo, na);
  k_cvt_hilo<<<cdiv(nw, 256), 256, 0, s>>>(W, sc.whi, sc.wlo, nw);
  launch_gemm_pre(sc.ahi, sc.alo, sc.whi, sc.wlo, bias, C, ldc, M, N, K, act, s);
}

// =====================================================================
// MHA core: softmax(QK^T/sqrt(80)) V  for seq-first tokens t = s*B + b
// QKV buffer is (S*B, 1920) with q/k/v at col offsets 0/640/1280.
// One block per (b, h, s-row); 128 threads.
// =====================================================================
__global__ void attn_kernel(const float* __restrict__ QKV,
                            float* __restrict__ O, int S, int B) {
  const int ld = 3 * CN;
  int idx = blockIdx.x;
  int s = idx % S; idx /= S;
  int h = idx % NHN; idx /= NHN;
  int b = idx;
  int tid = threadIdx.x;
  __shared__ float p[256];
  __shared__ float red[128];

  const float* q = QKV + (size_t)(s * B + b) * ld + h * HDN;
  const float scale = 0.1118033988749895f;  // 1/sqrt(80)

  for (int t = tid; t < S; t += 128) {
    const float* kk = QKV + (size_t)(t * B + b) * ld + CN + h * HDN;
    float acc = 0.f;
    #pragma unroll 4
    for (int d = 0; d < HDN; ++d) acc += q[d] * kk[d];
    p[t] = acc * scale;
  }
  __syncthreads();
  float m = -3.4e38f;
  for (int t = tid; t < S; t += 128) m = fmaxf(m, p[t]);
  red[tid] = m; __syncthreads();
  for (int w = 64; w > 0; w >>= 1) { if (tid < w) red[tid] = fmaxf(red[tid], red[tid + w]); __syncthreads(); }
  m = red[0]; __syncthreads();
  float ssum = 0.f;
  for (int t = tid; t < S; t += 128) { float e = __expf(p[t] - m); p[t] = e; ssum += e; }
  red[tid] = ssum; __syncthreads();
  for (int w = 64; w > 0; w >>= 1) { if (tid < w) red[tid] += red[tid + w]; __syncthreads(); }
  float inv = 1.0f / red[0];
  __syncthreads();
  for (int d = tid; d < HDN; d += 128) {
    float acc = 0.f;
    for (int t = 0; t < S; ++t)
      acc += p[t] * QKV[(size_t)(t * B + b) * ld + 2 * CN + h * HDN + d];
    O[(size_t)(s * B + b) * CN + h * HDN + d] = acc * inv;
  }
}

// =====================================================================
// LayerNorm over C=640:  y = LN(x (+res)) * g + b      (one block / row)
// =====================================================================
__global__ void ln_kernel(const float* __restrict__ x, const float* __restrict__ res,
                          const float* __restrict__ g, const float* __restrict__ bb,
                          float* __restrict__ y, int rows) {
  int r = blockIdx.x, tid = threadIdx.x;      // 256 threads
  __shared__ float red[256];
  float v[3]; int n = 0; float s = 0.f;
  for (int c = tid; c < CN; c += 256) {
    float t = x[(size_t)r * CN + c];
    if (res) t += res[(size_t)r * CN + c];
    v[n++] = t; s += t;
  }
  red[tid] = s; __syncthreads();
  for (int w = 128; w > 0; w >>= 1) { if (tid < w) red[tid] += red[tid + w]; __syncthreads(); }
  float mu = red[0] * (1.0f / CN); __syncthreads();
  s = 0.f;
  for (int i = 0; i < n; ++i) { float d = v[i] - mu; s += d * d; }
  red[tid] = s; __syncthreads();
  for (int w = 128; w > 0; w >>= 1) { if (tid < w) red[tid] += red[tid + w]; __syncthreads(); }
  float rs = rsqrtf(red[0] * (1.0f / CN) + 1e-5f);
  int i2 = 0;
  for (int c = tid; c < CN; c += 256)
    y[(size_t)r * CN + c] = (v[i2++] - mu) * rs * g[c] + bb[c];
}

// ---------------- small elementwise / data-movement kernels ----------------
__global__ void k_zero(float* p, int n) {
  int i = blockIdx.x * blockDim.x + threadIdx.x;
  if (i < n) p[i] = 0.f;
}
__global__ void k_addsc(float* o, const float* a, const float* b, float al, float be, int n) {
  int i = blockIdx.x * blockDim.x + threadIdx.x;
  if (i < n) o[i] = al * a[i] + be * b[i];
}
// (B,C,HW) -> token matrix (S*Bc, C), token t = s*Bc + bl, batch offset b0
__global__ void k_tok(const float* __restrict__ X, float* __restrict__ T, int Bc, int b0, int n) {
  int i = blockIdx.x * blockDim.x + threadIdx.x;
  if (i >= n) return;
  int c = i % CN; int t = i / CN; int bl = t % Bc; int s = t / Bc;
  T[i] = X[((size_t)(b0 + bl) * CN + c) * HWN + s];
}
// X_orig = 0.8*X + 0.2*attn_tokens  (undo token layout)
__global__ void k_blend(const float* __restrict__ X, const float* __restrict__ At,
                        float* __restrict__ Xo, int Bc, int b0, int n) {
  int i = blockIdx.x * blockDim.x + threadIdx.x;
  if (i >= n) return;
  int s = i % HWN; int c = (i / HWN) % CN; int bl = i / (HWN * CN);
  size_t gi = ((size_t)(b0 + bl) * CN + c) * HWN + s;
  Xo[gi] = 0.8f * X[gi] + 0.2f * At[(size_t)(s * Bc + bl) * CN + c];
}
// mean over HW (rows of (B,C,HW))
__global__ void pool_kernel(const float* __restrict__ X, float* __restrict__ P, int total) {
  int i = blockIdx.x * blockDim.x + threadIdx.x;
  if (i >= total) return;
  const float* p = X + (size_t)i * HWN;
  float s = 0.f;
  for (int j = 0; j < HWN; ++j) s += p[j];
  P[i] = s * (1.0f / HWN);
}

// =====================================================================
// Decoder step: s = softmax(qd kd^T) over WAY; score += s; tgt = s kd
// One block per query row (240 blocks, 128 threads)
// =====================================================================
__global__ void dec_kernel(const float* __restrict__ qd, const float* __restrict__ kd,
                           float* __restrict__ score_sum, float* __restrict__ tgt) {
  int q = blockIdx.x, tid = threadIdx.x;
  __shared__ float red[128];
  __shared__ float sm[16];
  int p = tid & 15, lane = tid >> 4;       // 8 lanes per prototype
  float acc = 0.f;
  for (int c = lane; c < CN; c += 8)
    acc += qd[(size_t)q * CN + c] * kd[(size_t)p * CN + c];
  red[tid] = acc; __syncthreads();
  if (tid < 16) {
    float s2 = 0.f;
    for (int j = 0; j < 8; ++j) s2 += red[p + 16 * j];
    sm[p] = s2;
  }
  __syncthreads();
  if (tid == 0) {
    float m = sm[0];
    for (int j = 1; j < 16; ++j) m = fmaxf(m, sm[j]);
    float s2 = 0.f;
    for (int j = 0; j < 16; ++j) { sm[j] = __expf(sm[j] - m); s2 += sm[j]; }
    float inv = 1.0f / s2;
    for (int j = 0; j < 16; ++j) sm[j] *= inv;
  }
  __syncthreads();
  if (tid < 16) score_sum[q * 16 + tid] += sm[tid];
  for (int c = tid; c < CN; c += 128) {
    float a = 0.f;
    for (int j = 0; j < 16; ++j) a += sm[j] * kd[(size_t)j * CN + c];
    tgt[(size_t)q * CN + c] = a;
  }
}

// =====================================================================
// Weight vectors: Wn[m,n,:] = normalize_hw( relu(Aimg[m,:,hw]·Bpool[n,:]) + 1e-3 )
// block per (m,n), 64 threads (thread j<49 owns pixel j)
// =====================================================================
__global__ void wv_kernel(const float* __restrict__ Aimg, const float* __restrict__ Bpool,
                          float* __restrict__ Wn, int N) {
  int m = blockIdx.x / N, n2 = blockIdx.x % N;
  int tid = threadIdx.x;
  __shared__ float w[HWN];
  __shared__ float red[64];
  if (tid < HWN) {
    const float* a = Aimg + (size_t)m * CN * HWN + tid;
    const float* bp = Bpool + (size_t)n2 * CN;
    float acc = 0.f;
    for (int c = 0; c < CN; ++c) acc += a[(size_t)c * HWN] * bp[c];
    w[tid] = fmaxf(acc, 0.f) + 0.001f;
  }
  __syncthreads();
  red[tid] = (tid < HWN) ? w[tid] : 0.f; __syncthreads();
  for (int s = 32; s > 0; s >>= 1) { if (tid < s) red[tid] += red[tid + s]; __syncthreads(); }
  float inv = 1.0f / red[0];
  if (tid < HWN) Wn[(size_t)blockIdx.x * HWN + tid] = w[tid] * inv;
}

// =====================================================================
// Channel-center + L2-normalize per pixel: Xn[(bi*49+j), c]
// block per (bi, j), 256 threads
// =====================================================================
__global__ void cn_kernel(const float* __restrict__ Ximg, float* __restrict__ Xn) {
  int bi = blockIdx.x / HWN, j = blockIdx.x % HWN;
  int tid = threadIdx.x;
  __shared__ float red[256];
  const float* x = Ximg + (size_t)bi * CN * HWN + j;
  float v[3]; int n = 0; float s = 0.f;
  for (int c = tid; c < CN; c += 256) { float t = x[(size_t)c * HWN]; v[n++] = t; s += t; }
  red[tid] = s; __syncthreads();
  for (int w = 128; w > 0; w >>= 1) { if (tid < w) red[tid] += red[tid + w]; __syncthreads(); }
  float mu = red[0] * (1.0f / CN); __syncthreads();
  s = 0.f;
  for (int i = 0; i < n; ++i) { float d = v[i] - mu; s += d * d; }
  red[tid] = s; __syncthreads();
  for (int w = 128; w > 0; w >>= 1) { if (tid < w) red[tid] += red[tid + w]; __syncthreads(); }
  float inv = 1.0f / fmaxf(sqrtf(red[0]), 1e-8f);
  __syncthreads();
  int i2 = 0;
  for (int c = tid; c < CN; c += 256)
    Xn[(size_t)blockIdx.x * CN + c] = (v[i2++] - mu) * inv;
}

// =====================================================================
// Factored EMD + final logits.
// weighted[q,p] = (Σ_m w1n qn[q,m,:]) · (Σ_n w2n pn[p,n,:])
// out[0:3840]=logits ; out[3840:7680]=score/TAU.   block per (q,p)
// =====================================================================
__global__ void emd_kernel(const float* __restrict__ qn, const float* __restrict__ pn,
                           const float* __restrict__ w1n, const float* __restrict__ w2n,
                           const float* __restrict__ score_sum, float* __restrict__ out) {
  int q = blockIdx.x >> 4, p = blockIdx.x & 15;
  int tid = threadIdx.x;
  __shared__ float sw1[HWN], sw2[HWN];
  __shared__ float red[256];
  if (tid < HWN) {
    sw1[tid] = w1n[(size_t)(q * 16 + p) * HWN + tid];
    sw2[tid] = w2n[(size_t)(p * 240 + q) * HWN + tid];
  }
  __syncthreads();
  float acc = 0.f;
  for (int c = tid; c < CN; c += 256) {
    float a = 0.f, b = 0.f;
    for (int m = 0; m < HWN; ++m) a += sw1[m] * qn[(size_t)(q * HWN + m) * CN + c];
    for (int n2 = 0; n2 < HWN; ++n2) b += sw2[n2] * pn[(size_t)(p * HWN + n2) * CN + c];
    acc += a * b;
  }
  red[tid] = acc; __syncthreads();
  for (int w = 128; w > 0; w >>= 1) { if (tid < w) red[tid] += red[tid + w]; __syncthreads(); }
  if (tid == 0) {
    float weighted = red[0];
    float sc = score_sum[blockIdx.x];
    out[blockIdx.x]        = 0.75f * weighted * (12.5f / 49.0f) + 0.25f * sc * 12.5f;
    out[3840 + blockIdx.x] = sc / 12.5f;
  }
}

// =====================================================================
// Host orchestration
// =====================================================================
extern "C" void kernel_launch(void* const* d_in, const int* in_sizes, int n_in,
                              void* d_out, int out_size, void* d_ws, size_t ws_size,
                              hipStream_t stream) {
  const float* support   = (const float*)d_in[0];   // (16,640,49)
  const float* query     = (const float*)d_in[1];   // (240,640,49)
  const float* patch_in_w  = (const float*)d_in[2];
  const float* patch_in_b  = (const float*)d_in[3];
  const float* patch_out_w = (const float*)d_in[4];
  const float* patch_out_b = (const float*)d_in[5];
  const float* enc_in_w  = (const float*)d_in[6];
  const float* enc_in_b  = (const float*)d_in[7];
  const float* enc_out_w = (const float*)d_in[8];
  const float* enc_out_b = (const float*)d_in[9];
  const float* enc_l1_w  = (const float*)d_in[10];
  const float* enc_l1_b  = (const float*)d_in[11];
  const float* enc_l2_w  = (const float*)d_in[12];
  const float* enc_l2_b  = (const float*)d_in[13];
  const float* enc_ln1_g = (const float*)d_in[14];
  const float* enc_ln1_b = (const float*)d_in[15];
  const float* enc_ln2_g = (const float*)d_in[16];
  const float* enc_ln2_b = (const float*)d_in[17];
  const float* sa_in_w   = (const float*)d_in[18];
  const float* sa_in_b   = (const float*)d_in[19];
  const float* sa_out_w  = (const float*)d_in[20];
  const float* sa_out_b  = (const float*)d_in[21];
  const float* dec_qkv_w = (const float*)d_in[22];
  const float* dec_qkv_b = (const float*)d_in[23];
  const float* dec_ln_g  = (const float*)d_in[24];
  const float* dec_ln_b  = (const float*)d_in[25];
  float* out = (float*)d_out;

  // ---- workspace layout (256B aligned) ----
  size_t off = 0;
  auto allocf = [&](size_t nfloats) -> float* {
    float* p = (float*)((char*)d_ws + off);
    off += (nfloats * sizeof(float) + 255) & ~(size_t)255;
    return p;
  };
  auto alloch = [&](size_t nhalves) -> _Float16* {
    _Float16* p = (_Float16*)((char*)d_ws + off);
    off += (nhalves * sizeof(_Float16) + 255) & ~(size_t)255;
    return p;
  };
  float* query_orig   = allocf((size_t)NQN * CN * HWN);
  float* support_orig = allocf((size_t)WAYN * CN * HWN);
  float* qn    = allocf((size_t)NQN * HWN * CN);
  float* pn    = allocf((size_t)WAYN * HWN * CN);
  float* tokbuf = allocf((size_t)QROWS * CN);
  float* qkvbuf = allocf((size_t)QROWS * 3 * CN);
  float* obuf   = allocf((size_t)QROWS * CN);
  float* abuf   = allocf((size_t)QROWS * CN);
  float* sup_pool = allocf((size_t)WAYN * CN);
  float* q_pool   = allocf((size_t)NQN * CN);
  float* spool_o  = allocf((size_t)WAYN * CN);
  float* qpool_o  = allocf((size_t)NQN * CN);
  float* qcur = allocf((size_t)NQN * CN);
  float* tgt  = allocf((size_t)NQN * CN);
  float* eqkv = allocf((size_t)WAYN * 3 * CN);
  float* eo   = allocf((size_t)WAYN * CN);
  float* eA   = allocf((size_t)WAYN * CN);
  float* x1   = allocf((size_t)WAYN * CN);
  float* ff1  = allocf((size_t)WAYN * FFNN);
  float* ff2  = allocf((size_t)WAYN * CN);
  float* memb = allocf((size_t)WAYN * CN);
  float* sqkv = allocf((size_t)NQN * 3 * CN);
  float* so   = allocf((size_t)NQN * CN);
  float* sA   = allocf((size_t)NQN * CN);
  float* tq   = allocf((size_t)NQN * CN);
  float* ln_tq  = allocf((size_t)NQN * CN);
  float* qd     = allocf((size_t)NQN * CN);
  float* ln_mem = allocf((size_t)WAYN * CN);
  float* kd     = allocf((size_t)WAYN * CN);
  float* w1n   = allocf((size_t)NQN * WAYN * HWN);
  float* w2n   = allocf((size_t)WAYN * NQN * HWN);
  float* score = allocf(NQN * WAYN);
  // f16 hi/lo planes
  GemmScratch sc;
  sc.ahi = alloch((size_t)QROWS * CN);           // largest A operand: 2352x640
  sc.alo = alloch((size_t)QROWS * CN);
  sc.whi = alloch((size_t)FFNN * CN);            // largest W operand: 2048x640
  sc.wlo = alloch((size_t)FFNN * CN);
  _Float16* pwin_hi  = alloch((size_t)3 * CN * CN);   // patch_in_w  1920x640
  _Float16* pwin_lo  = alloch((size_t)3 * CN * CN);
  _Float16* pwout_hi = alloch((size_t)CN * CN);       // patch_out_w 640x640
  _Float16* pwout_lo = alloch((size_t)CN * CN);
  (void)ws_size; (void)in_sizes; (void)n_in; (void)out_size;

  // convert shared patchFormer weights ONCE (reused by support + 5 chunks)
  k_cvt_hilo<<<cdiv(3 * CN * CN, 256), 256, 0, stream>>>(patch_in_w, pwin_hi, pwin_lo, 3 * CN * CN);
  k_cvt_hilo<<<cdiv(CN * CN, 256), 256, 0, stream>>>(patch_out_w, pwout_hi, pwout_lo, CN * CN);

  // ---------- phase 1: patchFormer MHA, support ----------
  {
    int T = HWN * WAYN;                                    // 784 rows
    int n = T * CN;
    k_tok<<<cdiv(n, 256), 256, 0, stream>>>(support, tokbuf, WAYN, 0, n);
    k_cvt_hilo<<<cdiv(n, 256), 256, 0, stream>>>(tokbuf, sc.ahi, sc.alo, n);
    launch_gemm_pre(sc.ahi, sc.alo, pwin_hi, pwin_lo, patch_in_b, qkvbuf, 3 * CN,
                    T, 3 * CN, CN, 0, stream);
    attn_kernel<<<WAYN * NHN * HWN, 128, 0, stream>>>(qkvbuf, obuf, HWN, WAYN);
    k_cvt_hilo<<<cdiv(n, 256), 256, 0, stream>>>(obuf, sc.ahi, sc.alo, n);
    launch_gemm_pre(sc.ahi, sc.alo, pwout_hi, pwout_lo, patch_out_b, abuf, CN,
                    T, CN, CN, 0, stream);
    int nb = WAYN * CN * HWN;
    k_blend<<<cdiv(nb, 256), 256, 0, stream>>>(support, abuf, support_orig, WAYN, 0, nb);
  }
  // ---------- phase 1: patchFormer MHA, query (chunks of 48) ----------
  for (int b0 = 0; b0 < NQN; b0 += QCH) {
    int T = QROWS;                                         // 2352 rows
    int n = T * CN;
    k_tok<<<cdiv(n, 256), 256, 0, stream>>>(query, tokbuf, QCH, b0, n);
    k_cvt_hilo<<<cdiv(n, 256), 256, 0, stream>>>(tokbuf, sc.ahi, sc.alo, n);
    launch_gemm_pre(sc.ahi, sc.alo, pwin_hi, pwin_lo, patch_in_b, qkvbuf, 3 * CN,
                    T, 3 * CN, CN, 0, stream);
    attn_kernel<<<QCH * NHN * HWN, 128, 0, stream>>>(qkvbuf, obuf, HWN, QCH);
    k_cvt_hilo<<<cdiv(n, 256), 256, 0, stream>>>(obuf, sc.ahi, sc.alo, n);
    launch_gemm_pre(sc.ahi, sc.alo, pwout_hi, pwout_lo, patch_out_b, abuf, CN,
                    T, CN, CN, 0, stream);
    int nb = QCH * CN * HWN;
    k_blend<<<cdiv(nb, 256), 256, 0, stream>>>(query, abuf, query_orig, QCH, b0, nb);
  }
  // ---------- pools ----------
  pool_kernel<<<cdiv(WAYN * CN, 256), 256, 0, stream>>>(support, sup_pool, WAYN * CN);
  pool_kernel<<<cdiv(NQN * CN, 256), 256, 0, stream>>>(query, q_pool, NQN * CN);
  pool_kernel<<<cdiv(WAYN * CN, 256), 256, 0, stream>>>(support_orig, spool_o, WAYN * CN);
  pool_kernel<<<cdiv(NQN * CN, 256), 256, 0, stream>>>(query_orig, qpool_o, NQN * CN);

  k_addsc<<<cdiv(NQN * CN, 256), 256, 0, stream>>>(qcur, q_pool, q_pool, 1.0f, 0.0f, NQN * CN);
  k_zero<<<cdiv(NQN * WAYN, 256), 256, 0, stream>>>(score, NQN * WAYN);

  // ---------- sampleFormer stack ----------
  for (int i = 0; i < 3; ++i) {
    if (i > 0)
      k_addsc<<<cdiv(NQN * CN, 256), 256, 0, stream>>>(qcur, qcur, tgt, 1.0f, 0.1f, NQN * CN);
    // encoder layer on prototypes (16 tokens)
    launch_gemm(sup_pool, enc_in_w + (size_t)i * 3 * CN * CN, enc_in_b + (size_t)i * 3 * CN,
                eqkv, 3 * CN, WAYN, 3 * CN, CN, 0, sc, stream);
    attn_kernel<<<NHN * WAYN, 128, 0, stream>>>(eqkv, eo, WAYN, 1);
    launch_gemm(eo, enc_out_w + (size_t)i * CN * CN, enc_out_b + (size_t)i * CN,
                eA, CN, WAYN, CN, CN, 0, sc, stream);
    ln_kernel<<<WAYN, 256, 0, stream>>>(eA, sup_pool, enc_ln1_g + (size_t)i * CN,
                                        enc_ln1_b + (size_t)i * CN, x1, WAYN);
    launch_gemm(x1, enc_l1_w + (size_t)i * FFNN * CN, enc_l1_b + (size_t)i * FFNN,
                ff1, FFNN, WAYN, FFNN, CN, 1, sc, stream);
    launch_gemm(ff1, enc_l2_w + (size_t)i * CN * FFNN, enc_l2_b + (size_t)i * CN,
                ff2, CN, WAYN, CN, FFNN, 0, sc, stream);
    ln_kernel<<<WAYN, 256, 0, stream>>>(ff2, x1, enc_ln2_g + (size_t)i * CN,
                                        enc_ln2_b + (size_t)i * CN, memb, WAYN);
    // self-attention on 240 query tokens
    launch_gemm(qcur, sa_in_w + (size_t)i * 3 * CN * CN, sa_in_b + (size_t)i * 3 * CN,
                sqkv, 3 * CN, NQN, 3 * CN, CN, 0, sc, stream);
    attn_kernel<<<NHN * NQN, 128, 0, stream>>>(sqkv, so, NQN, 1);
    launch_gemm(so, sa_out_w + (size_t)i * CN * CN, sa_out_b + (size_t)i * CN,
                sA, CN, NQN, CN, CN, 0, sc, stream);
    k_addsc<<<cdiv(NQN * CN, 256), 256, 0, stream>>>(tq, sA, qcur, 1.0f, 1.0f, NQN * CN);
    // decoder
    ln_kernel<<<NQN, 256, 0, stream>>>(tq, nullptr, dec_ln_g + (size_t)i * CN,
                                       dec_ln_b + (size_t)i * CN, ln_tq, NQN);
    launch_gemm(ln_tq, dec_qkv_w + (size_t)i * CN * CN, dec_qkv_b + (size_t)i * CN,
                qd, CN, NQN, CN, CN, 0, sc, stream);
    ln_kernel<<<WAYN, 256, 0, stream>>>(memb, nullptr, dec_ln_g + (size_t)i * CN,
                                        dec_ln_b + (size_t)i * CN, ln_mem, WAYN);
    launch_gemm(ln_mem, dec_qkv_w + (size_t)i * CN * CN, dec_qkv_b + (size_t)i * CN,
                kd, CN, WAYN, CN, CN, 0, sc, stream);
    dec_kernel<<<NQN, 128, 0, stream>>>(qd, kd, score, tgt);
  }

  // ---------- EMD branch (factored bilinear form) ----------
  wv_kernel<<<NQN * WAYN, 64, 0, stream>>>(query_orig, spool_o, w1n, WAYN);
  wv_kernel<<<WAYN * NQN, 64, 0, stream>>>(support_orig, qpool_o, w2n, NQN);
  cn_kernel<<<NQN * HWN, 256, 0, stream>>>(query_orig, qn);
  cn_kernel<<<WAYN * HWN, 256, 0, stream>>>(support_orig, pn);
  emd_kernel<<<NQN * WAYN, 256, 0, stream>>>(qn, pn, w1n, w2n, score, out);
}
